// GateActivation_70643622085237
// MI455X (gfx1250) — matrix operand order repaired and checked by reading, not compiled, not run on hardware
//
#include <hip/hip_runtime.h>
#include <math.h>
#include <stdint.h>

// ---------------------------------------------------------------------------
// MI455X (gfx1250) fused GateActivation forward.
//
// Roofline: 167.5 GFLOP vs ~1.0 GB HBM traffic (AI ~166 FLOP/B). HBM floor
// ~43us @23.3TB/s. fp32 VALU would be ~1ms -> must use WMMA. We use
// v_wmma_f32_16x16x32_bf16 with a 3-term hi/lo bf16 split (Markidis-style):
// xh*wh + xh*wl + xl*wh, dropped xl*wl ~2^-18 -> ~1e-5 rel error, fp32-grade,
// at ~3x bf16 WMMA issue (~100us est.).
//
// Round-2 changes:
//  * gemm_tile2: both 16-row blocks share one B-tile load (halves weight-pack
//    L2 traffic 7.2GB -> 3.6GB per pass; two independent WMMA acc chains).
//  * x-slab staging uses GLOBAL_LOAD_ASYNC_TO_LDS_B128 (ASYNCcnt) instead of
//    global->VGPR->ds_store (USE_ASYNC_LDS=0 reverts).
// ---------------------------------------------------------------------------

#ifndef USE_ASYNC_LDS
#define USE_ASYNC_LDS 1
#endif

typedef __attribute__((ext_vector_type(16))) __bf16         v16bf;
typedef __attribute__((ext_vector_type(16))) unsigned short v16u;
typedef __attribute__((ext_vector_type(8)))  float          v8f;

#define DEVINL static __device__ __forceinline__

DEVINL unsigned short f2bf(float f) {                  // fp32 -> bf16 (RNE)
  unsigned int u = __builtin_bit_cast(unsigned int, f);
  u += 0x7FFFu + ((u >> 16) & 1u);
  return (unsigned short)(u >> 16);
}
DEVINL float bf2f(unsigned short h) {
  unsigned int u = ((unsigned int)h) << 16;
  return __builtin_bit_cast(float, u);
}
DEVINL float sigmoidf_(float x) { return 1.0f / (1.0f + __expf(-x)); }

// ---- weight-pack layout in d_ws (ushort elements) ----
constexpr size_t SZ_W0PRE  = 512 * 896;
constexpr size_t SZ_W1PRE  = 256 * 256;
constexpr size_t SZ_W2PRE  = 128 * 128;
constexpr size_t SZ_W0POST = 512 * 512;
constexpr size_t SZ_W1POST = 256 * 256;
constexpr size_t SZ_W2POST = 128 * 128;
constexpr size_t OFF_W0PRE_HI  = 0;
constexpr size_t OFF_W0PRE_LO  = OFF_W0PRE_HI  + SZ_W0PRE;
constexpr size_t OFF_W1PRE_HI  = OFF_W0PRE_LO  + SZ_W0PRE;
constexpr size_t OFF_W1PRE_LO  = OFF_W1PRE_HI  + SZ_W1PRE;
constexpr size_t OFF_W2PRE_HI  = OFF_W1PRE_LO  + SZ_W1PRE;
constexpr size_t OFF_W2PRE_LO  = OFF_W2PRE_HI  + SZ_W2PRE;
constexpr size_t OFF_W0POST_HI = OFF_W2PRE_LO  + SZ_W2PRE;
constexpr size_t OFF_W0POST_LO = OFF_W0POST_HI + SZ_W0POST;
constexpr size_t OFF_W1POST_HI = OFF_W0POST_LO + SZ_W0POST;
constexpr size_t OFF_W1POST_LO = OFF_W1POST_HI + SZ_W1POST;
constexpr size_t OFF_W2POST_HI = OFF_W1POST_LO + SZ_W1POST;
constexpr size_t OFF_W2POST_LO = OFF_W2POST_HI + SZ_W2POST;
// total 1,769,472 ushorts = 3,538,944 bytes

// ---------------------------------------------------------------------------
// Prep: W[K x N] f32 row-major -> hi/lo bf16 B-operand tiles (32k x 16n).
// B layout (v_wmma_*_16x16x32 B = 32x16): lane<16: col=lane,    j<8 -> k=j,   j>=8 -> k=j+8;
//                                         lane>=16: col=lane-16, j<8 -> k=j+8, j>=8 -> k=j+16.
// Pack element e = ((tk*NT + tn)*32 + lane)*16 + j  (contiguous per lane).
// ---------------------------------------------------------------------------
__global__ void prep_pack(const float* __restrict__ W, int K, int N, float scale,
                          unsigned short* __restrict__ hi, unsigned short* __restrict__ lo)
{
  int e = blockIdx.x * 256 + threadIdx.x;
  if (e >= K * N) return;
  int NT = N >> 4;
  int j  = e & 15;
  int lp = (e >> 4) & 31;
  int t  = e >> 9;
  int tn = t % NT, tk = t / NT;
  int k  = tk * 32 + (j & 7) + ((j >> 3) << 4) + ((lp >> 4) << 3);
  int n  = tn * 16 + (lp & 15);
  float v = W[(size_t)k * N + n] * scale;
  unsigned short h = f2bf(v);
  hi[e] = h;
  lo[e] = f2bf(v - bf2f(h));
}

// ---------------------------------------------------------------------------
// A-operand pack helper. A layout (16x32 bf16, per ISA table):
//   lane<16: row=lane;     j<8 -> k=j,    j>=8 -> k=j+8
//   lane>=16: row=lane-16; j<8 -> k=j+8,  j>=8 -> k=j+16
// ---------------------------------------------------------------------------
DEVINL void packA_store(unsigned short* dH, unsigned short* dL, int m, int k, float v)
{
  int tk = k >> 5;
  int lp = (m & 15) + (((k >> 3) & 1) << 4);
  int j  = (k & 7)  + (((k >> 4) & 1) << 3);
  int e  = tk * 512 + lp * 16 + j;
  unsigned short h = f2bf(v);
  dH[e] = h;
  dL[e] = f2bf(v - bf2f(h));
}

#define WMMA_BF16(A, B, C) \
  __builtin_amdgcn_wmma_f32_16x16x32_bf16(false, (A), false, (B), (short)0, (C), false, false)

// 3-term split-bf16 GEMM over a 32x16 output strip (both rowblocks at once so
// each B tile is loaded from global exactly once; two independent acc chains).
DEVINL void gemm_tile2(const unsigned short* aH, const unsigned short* aL, int rbs,
                       const unsigned short* bHi, const unsigned short* bLo,
                       int tn, int NT, int nTK, int lane, v8f& acc0, v8f& acc1)
{
  const int aoff = lane * 16;
#pragma unroll 4
  for (int tk = 0; tk < nTK; ++tk) {
    size_t bo = ((size_t)(tk * NT + tn) * 32 + lane) * 16;
    v16bf B   = __builtin_bit_cast(v16bf, *(const v16u*)(bHi + bo));
    v16bf Bl  = __builtin_bit_cast(v16bf, *(const v16u*)(bLo + bo));
    const unsigned short* a0 = aH + tk * 512 + aoff;
    const unsigned short* l0 = aL + tk * 512 + aoff;
    v16bf A0  = __builtin_bit_cast(v16bf, *(const v16u*)(a0));
    v16bf A0l = __builtin_bit_cast(v16bf, *(const v16u*)(l0));
    v16bf A1  = __builtin_bit_cast(v16bf, *(const v16u*)(a0 + rbs));
    v16bf A1l = __builtin_bit_cast(v16bf, *(const v16u*)(l0 + rbs));
    acc0 = WMMA_BF16(A0,  B,  acc0);
    acc0 = WMMA_BF16(A0,  Bl, acc0);
    acc0 = WMMA_BF16(A0l, B,  acc0);
    acc1 = WMMA_BF16(A1,  B,  acc1);
    acc1 = WMMA_BF16(A1,  Bl, acc1);
    acc1 = WMMA_BF16(A1l, B,  acc1);
  }
}

// ---------------------------------------------------------------------------
// Stage a 32-row slab of x[:, colbase : colbase+COLS] into LDS (contiguous,
// row stride COLS). Async direct-to-LDS path avoids the VGPR round trip and
// is tracked with ASYNCcnt (s_wait_asynccnt before the barrier).
// ---------------------------------------------------------------------------
template <int COLS>
DEVINL void stage_slab(const float* __restrict__ x, int row0, int colbase,
                       float* rawf, int tid)
{
  constexpr int C4 = COLS >> 2;             // 16B chunks per row
#if USE_ASYNC_LDS
  for (int idx = tid; idx < 32 * C4; idx += 256) {
    int r = idx / C4, c4 = idx - r * C4;
    const float* g = x + (size_t)(row0 + r) * 1920 + colbase + (c4 << 2);
    unsigned lds = (unsigned)(uintptr_t)(rawf + r * COLS + (c4 << 2));
    asm volatile("global_load_async_to_lds_b128 %0, %1, off"
                 :: "v"(lds), "v"((unsigned long long)(uintptr_t)g)
                 : "memory");
  }
  asm volatile("s_wait_asynccnt 0" ::: "memory");
#else
  for (int idx = tid; idx < 32 * C4; idx += 256) {
    int r = idx / C4, c4 = idx - r * C4;
    const float4* g = (const float4*)(x + (size_t)(row0 + r) * 1920 + colbase) + c4;
    ((float4*)(rawf + r * COLS))[c4] = *g;
  }
#endif
}

// ---------------------------------------------------------------------------
// Fused kernel: 32 rows / workgroup, 256 threads (8 wave32 waves).
// LDS regions (240 KB of the 320 KB WGP budget):
//   [0,96K)     region A : raw f32 slab, later stage-2 A packs (hi|lo)
//   [96K,192K)  region B : stage-1 A packs (hi|lo)
//   [192K,240K) gate     : g[32][384] f32
// ---------------------------------------------------------------------------
__global__ __launch_bounds__(256) void fused_forward(const float* __restrict__ x,
                                                     float* __restrict__ out,
                                                     const unsigned short* __restrict__ wp)
{
  __shared__ __align__(16) unsigned char smem[245760];
  float*          rawf = (float*)smem;
  unsigned short* A2   = (unsigned short*)smem;
  unsigned short* A1   = (unsigned short*)(smem + 98304);
  float*          gate = (float*)(smem + 196608);

  const int tid  = threadIdx.x;
  const int wid  = tid >> 5;
  const int lane = tid & 31;
  const int row0 = blockIdx.x * 32;

  const int laneN = lane & 15;           // output-tile column within tile
  const int mbh   = (lane >> 4) << 3;    // +8 row offset for lanes 16..31

  // ======================= Phase S =========================================
  stage_slab<512>(x, row0, 0, rawf, tid);
  __syncthreads();
  // pack A1s: [rb][tk<16][lp][j], hi at 0, lo at +16384
  for (int e = tid; e < 16384; e += 256) {
    int j = e & 15, lp = (e >> 4) & 31, tk = (e >> 9) & 15, rb = e >> 13;
    int kk = (j & 7) + ((j >> 3) << 4) + ((lp >> 4) << 3);
    float v = rawf[(rb * 16 + (lp & 15)) * 512 + tk * 32 + kk];
    unsigned short h = f2bf(v);
    A1[e]         = h;
    A1[16384 + e] = f2bf(v - bf2f(h));
  }
  __syncthreads();
  // G0pre: [32x512] @ W0_pre[512x896], NT=56, nTK=16
  for (int ci = 0; ci < 7; ++ci) {
    int c = wid + ci * 8;
    v8f acc[2] = {{0.f,0.f,0.f,0.f,0.f,0.f,0.f,0.f},
                  {0.f,0.f,0.f,0.f,0.f,0.f,0.f,0.f}};
    gemm_tile2(A1, A1 + 16384, 8192,
               wp + OFF_W0PRE_HI, wp + OFF_W0PRE_LO, c, 56, 16, lane, acc[0], acc[1]);
    int colb = c * 16 + laneN;
    if (c < 32) {                       // silu -> stage-2 A pack (region A)
#pragma unroll
      for (int rb = 0; rb < 2; ++rb) {
        int mb = rb * 16 + mbh;
        unsigned short* dH = A2 + rb * 8192;
        unsigned short* dL = A2 + 16384 + rb * 8192;
#pragma unroll
        for (int r = 0; r < 8; ++r) {
          float v = acc[rb][r];
          packA_store(dH, dL, mb + r, colb, v * sigmoidf_(v));
        }
      }
    } else {                            // sigmoid gate -> LDS f32
#pragma unroll
      for (int rb = 0; rb < 2; ++rb) {
        int mb = rb * 16 + mbh;
#pragma unroll
        for (int r = 0; r < 8; ++r)
          gate[(mb + r) * 384 + (colb - 512)] = sigmoidf_(acc[rb][r]);
      }
    }
  }
  __syncthreads();
  // G0post: silu(s) @ W0_post[512x512], NT=32, nTK=16
  for (int ci = 0; ci < 4; ++ci) {
    int c = wid + ci * 8;
    v8f acc[2] = {{0.f,0.f,0.f,0.f,0.f,0.f,0.f,0.f},
                  {0.f,0.f,0.f,0.f,0.f,0.f,0.f,0.f}};
    gemm_tile2(A2, A2 + 16384, 8192,
               wp + OFF_W0POST_HI, wp + OFF_W0POST_LO, c, 32, 16, lane, acc[0], acc[1]);
    int col = c * 16 + laneN;
#pragma unroll
    for (int rb = 0; rb < 2; ++rb) {
      int mb = rb * 16 + mbh;
#pragma unroll
      for (int r = 0; r < 8; ++r)
        out[(size_t)(row0 + mb + r) * 1920 + col] = acc[rb][r];
    }
  }
  __syncthreads();

  // ======================= Phase V1 ========================================
  stage_slab<768>(x, row0, 512, rawf, tid);
  __syncthreads();
  // pack 3 channels: ch block = i*16384 (hi 8192 | lo 8192), per-rb 4096 (nTK=8)
  for (int e2 = tid; e2 < 3 * 8192; e2 += 256) {
    int i = e2 >> 13, e = e2 & 8191;
    int j = e & 15, lp = (e >> 4) & 31, tk = (e >> 9) & 7, rb = e >> 12;
    int kk = (j & 7) + ((j >> 3) << 4) + ((lp >> 4) << 3);
    float v = rawf[(rb * 16 + (lp & 15)) * 768 + (tk * 32 + kk) * 3 + i];
    unsigned short h = f2bf(v);
    A1[i * 16384 + e]        = h;
    A1[i * 16384 + 8192 + e] = f2bf(v - bf2f(h));
  }
  __syncthreads();
  // G1pre: per channel [32x256] @ W1_pre[256x256]; gate; repack to region A
  for (int ci = 0; ci < 6; ++ci) {
    int cc = wid + ci * 8, i = cc >> 4, tn = cc & 15;
    v8f acc[2] = {{0.f,0.f,0.f,0.f,0.f,0.f,0.f,0.f},
                  {0.f,0.f,0.f,0.f,0.f,0.f,0.f,0.f}};
    gemm_tile2(A1 + i * 16384, A1 + i * 16384 + 8192, 4096,
               wp + OFF_W1PRE_HI, wp + OFF_W1PRE_LO, tn, 16, 8, lane, acc[0], acc[1]);
    int vcol = tn * 16 + laneN;
#pragma unroll
    for (int rb = 0; rb < 2; ++rb) {
      int mb = rb * 16 + mbh;
      unsigned short* dH = A2 + i * 16384 + rb * 4096;
      unsigned short* dL = A2 + i * 16384 + 8192 + rb * 4096;
#pragma unroll
      for (int r = 0; r < 8; ++r) {
        float v = acc[rb][r] * gate[(mb + r) * 384 + vcol];
        packA_store(dH, dL, mb + r, vcol, v);
      }
    }
  }
  __syncthreads();
  // G1post: gated v1 @ W1_post; store out[:, 512 + v*3 + i]
  for (int ci = 0; ci < 6; ++ci) {
    int cc = wid + ci * 8, i = cc >> 4, tn = cc & 15;
    v8f acc[2] = {{0.f,0.f,0.f,0.f,0.f,0.f,0.f,0.f},
                  {0.f,0.f,0.f,0.f,0.f,0.f,0.f,0.f}};
    gemm_tile2(A2 + i * 16384, A2 + i * 16384 + 8192, 4096,
               wp + OFF_W1POST_HI, wp + OFF_W1POST_LO, tn, 16, 8, lane, acc[0], acc[1]);
    int vcol = tn * 16 + laneN;
#pragma unroll
    for (int rb = 0; rb < 2; ++rb) {
      int mb = rb * 16 + mbh;
#pragma unroll
      for (int r = 0; r < 8; ++r)
        out[(size_t)(row0 + mb + r) * 1920 + 512 + vcol * 3 + i] = acc[rb][r];
    }
  }
  __syncthreads();

  // ======================= Phase V2 ========================================
  stage_slab<640>(x, row0, 1280, rawf, tid);
  __syncthreads();
  // pack 5 channels: ch block = i*8192 (hi 4096 | lo 4096), per-rb 2048 (nTK=4)
  for (int e2 = tid; e2 < 5 * 4096; e2 += 256) {
    int i = e2 >> 12, e = e2 & 4095;
    int j = e & 15, lp = (e >> 4) & 31, tk = (e >> 9) & 3, rb = e >> 11;
    int kk = (j & 7) + ((j >> 3) << 4) + ((lp >> 4) << 3);
    float v = rawf[(rb * 16 + (lp & 15)) * 640 + (tk * 32 + kk) * 5 + i];
    unsigned short h = f2bf(v);
    A1[i * 8192 + e]        = h;
    A1[i * 8192 + 4096 + e] = f2bf(v - bf2f(h));
  }
  __syncthreads();
  // G2pre: per channel [32x128] @ W2_pre[128x128]; gate (cols 256..383); repack
  for (int ci = 0; ci < 5; ++ci) {
    int cc = wid + ci * 8, i = cc >> 3, tn = cc & 7;
    v8f acc[2] = {{0.f,0.f,0.f,0.f,0.f,0.f,0.f,0.f},
                  {0.f,0.f,0.f,0.f,0.f,0.f,0.f,0.f}};
    gemm_tile2(A1 + i * 8192, A1 + i * 8192 + 4096, 2048,
               wp + OFF_W2PRE_HI, wp + OFF_W2PRE_LO, tn, 8, 4, lane, acc[0], acc[1]);
    int vcol = tn * 16 + laneN;
#pragma unroll
    for (int rb = 0; rb < 2; ++rb) {
      int mb = rb * 16 + mbh;
      unsigned short* dH = A2 + i * 8192 + rb * 2048;
      unsigned short* dL = A2 + i * 8192 + 4096 + rb * 2048;
#pragma unroll
      for (int r = 0; r < 8; ++r) {
        float v = acc[rb][r] * gate[(mb + r) * 384 + 256 + vcol];
        packA_store(dH, dL, mb + r, vcol, v);
      }
    }
  }
  __syncthreads();
  // G2post: gated v2 @ W2_post; store out[:, 1280 + v*5 + i]
  for (int ci = 0; ci < 5; ++ci) {
    int cc = wid + ci * 8, i = cc >> 3, tn = cc & 7;
    v8f acc[2] = {{0.f,0.f,0.f,0.f,0.f,0.f,0.f,0.f},
                  {0.f,0.f,0.f,0.f,0.f,0.f,0.f,0.f}};
    gemm_tile2(A2 + i * 8192, A2 + i * 8192 + 4096, 2048,
               wp + OFF_W2POST_HI, wp + OFF_W2POST_LO, tn, 8, 4, lane, acc[0], acc[1]);
    int vcol = tn * 16 + laneN;
#pragma unroll
    for (int rb = 0; rb < 2; ++rb) {
      int mb = rb * 16 + mbh;
#pragma unroll
      for (int r = 0; r < 8; ++r)
        out[(size_t)(row0 + mb + r) * 1920 + 1280 + vcol * 5 + i] = acc[rb][r];
    }
  }
}

// ---------------------------------------------------------------------------
extern "C" void kernel_launch(void* const* d_in, const int* in_sizes, int n_in,
                              void* d_out, int out_size, void* d_ws, size_t ws_size,
                              hipStream_t stream) {
  (void)in_sizes; (void)n_in; (void)out_size; (void)ws_size;
  const float* x       = (const float*)d_in[0];
  const float* W0_pre  = (const float*)d_in[1];
  const float* W1_pre  = (const float*)d_in[2];
  const float* W2_pre  = (const float*)d_in[3];
  const float* W0_post = (const float*)d_in[4];
  const float* W1_post = (const float*)d_in[5];
  const float* W2_post = (const float*)d_in[6];
  unsigned short* wp   = (unsigned short*)d_ws;   // needs 3,538,944 bytes

  const float s512 = 1.0f / sqrtf(512.0f);
  const float s256 = 1.0f / sqrtf(256.0f);
  const float s128 = 1.0f / sqrtf(128.0f);

  auto prep = [&](const float* W, int K, int N, float sc, size_t oh, size_t ol) {
    int total = K * N;
    prep_pack<<<(total + 255) / 256, 256, 0, stream>>>(W, K, N, sc, wp + oh, wp + ol);
  };
  prep(W0_pre,  512, 896, s512, OFF_W0PRE_HI,  OFF_W0PRE_LO);
  prep(W1_pre,  256, 256, s256, OFF_W1PRE_HI,  OFF_W1PRE_LO);
  prep(W2_pre,  128, 128, s128, OFF_W2PRE_HI,  OFF_W2PRE_LO);
  prep(W0_post, 512, 512, s512, OFF_W0POST_HI, OFF_W0POST_LO);
  prep(W1_post, 256, 256, s256, OFF_W1POST_HI, OFF_W1POST_LO);
  prep(W2_post, 128, 128, s128, OFF_W2POST_HI, OFF_W2POST_LO);

  fused_forward<<<65536 / 32, 256, 0, stream>>>(x, (float*)d_out, wp);
}
